// AlternateAttention_73486890434741
// MI455X (gfx1250) — matrix operand order repaired and checked by reading, not compiled, not run on hardware
//
#include <hip/hip_runtime.h>
#include <hip/hip_bf16.h>

#define BB 2
#define SS 4096
#define EE 512
#define HH 8
#define HDD 64
#define WBAND 16
#define FFF 2048
#define MM (BB * SS)   // 8192 tokens

typedef __attribute__((ext_vector_type(16))) _Float16 v16h;
typedef __attribute__((ext_vector_type(8)))  _Float16 v8h;
typedef __attribute__((ext_vector_type(8)))  float    v8f;

// ---------------------------------------------------------------------------
// WMMA helpers
// ---------------------------------------------------------------------------
__device__ __forceinline__ v8f wmma32(v16h a, v16h b, v8f c) {
  // D = A(16x32 f16) x B(32x16 f16) + C(16x16 f32)
  return __builtin_amdgcn_wmma_f32_16x16x32_f16(
      /*neg_a=*/false, a, /*neg_b=*/false, b,
      /*c_mod=*/(short)0, c, /*reuse_a=*/false, /*reuse_b=*/false);
}

// A-matrix 16x32 f16 per-lane load pattern: lane holds row (lane&15),
// elements 0..15 = K kb+0..15, kb = 16*(lane>>4).
// Caller passes base already offset by kb.
__device__ __forceinline__ v16h ldA(const _Float16* p) {
  v8h lo = *(const v8h*)(p);
  v8h hi = *(const v8h*)(p + 8);
  v16h r;
#pragma unroll
  for (int i = 0; i < 8; ++i) { r[i] = lo[i]; r[i + 8] = hi[i]; }
  return r;
}
// B-matrix 32x16 f16: lane holds column (lane&15); elements are 16 contiguous
// K values starting at 16*(lane>>4). Single aligned 32-byte load.

__device__ __forceinline__ float gelu_tanh(float x) {
  float x3 = x * x * x;
  return 0.5f * x * (1.0f + tanhf(0.7978845608028654f * (x + 0.044715f * x3)));
}

// ---------------------------------------------------------------------------
// f32 -> f16 convert (weights)
// ---------------------------------------------------------------------------
__global__ void __launch_bounds__(256) cvt_f16_kernel(
    const float* __restrict__ in, _Float16* __restrict__ out, int n) {
  int i = blockIdx.x * 256 + threadIdx.x;
  if (i < n) out[i] = (_Float16)in[i];
}

// ---------------------------------------------------------------------------
// LayerNorm (E=512) -> f16.  One 256-thread block per token, 2 elems/thread.
// ---------------------------------------------------------------------------
__global__ void __launch_bounds__(256) ln_f16_kernel(
    const float* __restrict__ x, const float* __restrict__ w,
    const float* __restrict__ b, _Float16* __restrict__ out) {
  int tok = blockIdx.x;
  const float* xp = x + (size_t)tok * EE;
  int t = threadIdx.x;
  float v0 = xp[t], v1 = xp[t + 256];
  float s = v0 + v1, s2 = v0 * v0 + v1 * v1;
#pragma unroll
  for (int off = 16; off >= 1; off >>= 1) {
    s  += __shfl_xor(s,  off);
    s2 += __shfl_xor(s2, off);
  }
  __shared__ float sh[16];
  int wv = t >> 5;
  if ((t & 31) == 0) { sh[wv] = s; sh[wv + 8] = s2; }
  __syncthreads();
  float ts = 0.f, ts2 = 0.f;
#pragma unroll
  for (int i = 0; i < 8; ++i) { ts += sh[i]; ts2 += sh[i + 8]; }
  float mean = ts * (1.0f / EE);
  float var  = ts2 * (1.0f / EE) - mean * mean;
  float inv  = rsqrtf(var + 1e-5f);
  _Float16* op = out + (size_t)tok * EE;
  op[t]       = (_Float16)((v0 - mean) * inv * w[t]       + b[t]);
  op[t + 256] = (_Float16)((v1 - mean) * inv * w[t + 256] + b[t + 256]);
}

// ---------------------------------------------------------------------------
// WMMA GEMM: Y[M,N] = A16[M,K] @ W16[N,K]^T + bias, fused epilogues.
// Each wave computes a 32x64 output tile: 2 M-subtiles x 4 N-subtiles,
// 8 accumulators.  A loads amortized 4x, B loads amortized 2x -> 8 WMMAs
// per 12 16B-loads per K-step (compute-bound regime, maximize wmma density).
// ---------------------------------------------------------------------------
enum { EPI_QKV = 0, EPI_GELU = 1, EPI_RESID = 2 };

template <int MODE>
__global__ void __launch_bounds__(256) gemm_wmma_kernel(
    const _Float16* __restrict__ A, const _Float16* __restrict__ Wt,
    const float* __restrict__ bias,
    float* fout, const float* resid,          // EPI_RESID (may alias)
    _Float16* __restrict__ hout,              // EPI_GELU
    _Float16* __restrict__ qo, _Float16* __restrict__ ko,
    _Float16* __restrict__ vo,                // EPI_QKV
    int M, int N, int K) {
  int wv = threadIdx.x >> 5, lane = threadIdx.x & 31;
  int gw = blockIdx.x * 8 + wv;
  int mtiles = M >> 5, ntiles4 = N >> 6;   // 32-row x 64-col wave tiles
  if (gw >= mtiles * ntiles4) return;
  int mt = gw % mtiles, n4 = gw / mtiles;
  int lrow = lane & 15;
  int hi   = lane >> 4;
  int lka  = hi * 16;   // A K-offset (halves)
  int lkb  = hi * 16;   // B K-offset (halves)

  const _Float16* arow0 = A + (size_t)(mt * 32 + lrow) * K + lka;
  const _Float16* arow1 = arow0 + (size_t)16 * K;
  const _Float16* brow[4];
#pragma unroll
  for (int s4 = 0; s4 < 4; ++s4)
    brow[s4] = Wt + (size_t)(n4 * 64 + s4 * 16 + lrow) * K + lkb;

  v8f acc[2][4] = {};
  for (int k = 0; k < K; k += 32) {
    v16h a0 = ldA(arow0 + k);
    v16h a1 = ldA(arow1 + k);
    __builtin_prefetch(arow0 + k + 256, 0, 0);
    __builtin_prefetch(arow1 + k + 256, 0, 0);
#pragma unroll
    for (int s4 = 0; s4 < 4; ++s4) {
      v16h b = *(const v16h*)(brow[s4] + k);
      acc[0][s4] = wmma32(a0, b, acc[0][s4]);
      acc[1][s4] = wmma32(a1, b, acc[1][s4]);
    }
  }

  // Epilogue (C layout: element r -> row r+8*hi, col lane&15)
#pragma unroll
  for (int mi = 0; mi < 2; ++mi) {
#pragma unroll
    for (int s4 = 0; s4 < 4; ++s4) {
      int col = n4 * 64 + s4 * 16 + lrow;
      float bv = bias[col];
#pragma unroll
      for (int r = 0; r < 8; ++r) {
        int row = mt * 32 + mi * 16 + r + 8 * hi;
        float v = acc[mi][s4][r] + bv;
        if (MODE == EPI_GELU) {
          hout[(size_t)row * N + col] = (_Float16)gelu_tanh(v);
        } else if (MODE == EPI_RESID) {
          size_t idx = (size_t)row * N + col;
          fout[idx] = resid[idx] + v;
        } else {  // EPI_QKV: split into per-head Q,K (row-major) and V^T
          int b   = row >> 12;          // / SS
          int s   = row & (SS - 1);
          int prt = col >> 9;           // 0=q,1=k,2=v
          int c   = col & (EE - 1);
          int h   = c >> 6, d = c & (HDD - 1);
          if (prt == 0)
            qo[((size_t)(b * HH + h) * SS + s) * HDD + d] = (_Float16)v;
          else if (prt == 1)
            ko[((size_t)(b * HH + h) * SS + s) * HDD + d] = (_Float16)v;
          else
            vo[((size_t)(b * HH + h) * HDD + d) * SS + s] = (_Float16)v;
        }
      }
    }
  }
}

// ---------------------------------------------------------------------------
// Flash attention, 16-query tile per wave, 32-key blocks.
// mode 0 = global, 1 = lower band (causal width 16), 2 = upper band.
// ---------------------------------------------------------------------------
__device__ __forceinline__ float mbias(int mode, int i, int j) {
  if (mode == 1) return (j <= i && j >= i - (WBAND - 1)) ? 0.0f : -1e30f;
  if (mode == 2) return (j >= i && j < i + WBAND) ? 0.0f : -1e30f;
  return 0.0f;
}

__global__ void __launch_bounds__(256) flash_attn_kernel(
    const _Float16* __restrict__ q, const _Float16* __restrict__ kk,
    const _Float16* __restrict__ vt, _Float16* __restrict__ o, int mode) {
  __shared__ _Float16 Pl[8][16 * 32];  // per-wave P tile staging
  int wv = threadIdx.x >> 5, lane = threadIdx.x & 31;
  int gw = blockIdx.x * 8 + wv;        // B*H*(S/16) = 4096 waves total
  int qt = gw & (SS / 16 - 1);
  int bh = gw >> 8;                    // b*H + h
  int lrow = lane & 15, hi = lane >> 4;
  int lka = hi * 16, lkb = hi * 16;

  const _Float16* qb = q  + (size_t)bh * SS * HDD;
  const _Float16* kb = kk + (size_t)bh * SS * HDD;
  const _Float16* vb = vt + (size_t)bh * HDD * SS;

  v16h qa0 = ldA(qb + (size_t)(qt * 16 + lrow) * HDD + lka);
  v16h qa1 = ldA(qb + (size_t)(qt * 16 + lrow) * HDD + 32 + lka);

  v8f oacc[4] = {};
  float m[8], l[8];
#pragma unroll
  for (int r = 0; r < 8; ++r) { m[r] = -1e20f; l[r] = 0.0f; }

  int jb0, jb1;
  if (mode == 0) { jb0 = 0; jb1 = SS - 32; }
  else if (mode == 1) {
    int lo = qt * 16 - (WBAND - 1); if (lo < 0) lo = 0;
    jb0 = lo & ~31;
    jb1 = (qt * 16 + 15) & ~31;
  } else {
    jb0 = (qt * 16) & ~31;
    int hj = qt * 16 + 15 + (WBAND - 1); if (hj > SS - 1) hj = SS - 1;
    jb1 = hj & ~31;
  }

  for (int jb = jb0; jb <= jb1; jb += 32) {
    // --- QK^T: two 16x16 score tiles over this 32-key block ---
    v8f c0 = {}, c1 = {};
    {
      const _Float16* kr0 = kb + (size_t)(jb + lrow) * HDD + lkb;
      const _Float16* kr1 = kb + (size_t)(jb + 16 + lrow) * HDD + lkb;
      c0 = wmma32(qa0, *(const v16h*)(kr0),      c0);
      c0 = wmma32(qa1, *(const v16h*)(kr0 + 32), c0);
      c1 = wmma32(qa0, *(const v16h*)(kr1),      c1);
      c1 = wmma32(qa1, *(const v16h*)(kr1 + 32), c1);
    }
    float s0[8], s1[8], rm[8];
#pragma unroll
    for (int r = 0; r < 8; ++r) {
      int i = qt * 16 + r + 8 * hi;
      s0[r] = c0[r] * 0.125f + mbias(mode, i, jb + lrow);
      s1[r] = c1[r] * 0.125f + mbias(mode, i, jb + 16 + lrow);
      float v = fmaxf(s0[r], s1[r]);          // row max across 16 columns
      v = fmaxf(v, __shfl_xor(v, 1));
      v = fmaxf(v, __shfl_xor(v, 2));
      v = fmaxf(v, __shfl_xor(v, 4));
      v = fmaxf(v, __shfl_xor(v, 8));
      rm[r] = v;
    }
#pragma unroll
    for (int r = 0; r < 8; ++r) {
      float mn = fmaxf(m[r], rm[r]);          // floored at -1e20 via init
      float sc = __expf(m[r] - mn);
      float p0 = __expf(s0[r] - mn);
      float p1 = __expf(s1[r] - mn);
      float rs = p0 + p1;
      rs += __shfl_xor(rs, 1);
      rs += __shfl_xor(rs, 2);
      rs += __shfl_xor(rs, 4);
      rs += __shfl_xor(rs, 8);
      l[r] = l[r] * sc + rs;
      m[r] = mn;
#pragma unroll
      for (int s4 = 0; s4 < 4; ++s4) oacc[s4][r] *= sc;
      Pl[wv][(r + 8 * hi) * 32 + lrow]      = (_Float16)p0;
      Pl[wv][(r + 8 * hi) * 32 + 16 + lrow] = (_Float16)p1;
    }
    asm volatile("s_wait_dscnt 0" ::: "memory");
    // Re-read P in A-matrix layout
    v16h pa;
#pragma unroll
    for (int i2 = 0; i2 < 8; ++i2) {
      pa[i2]     = Pl[wv][lrow * 32 + lka + i2];
      pa[i2 + 8] = Pl[wv][lrow * 32 + lka + 8 + i2];
    }
    // --- P @ V over HD=64 (V stored transposed [HD,S]) ---
#pragma unroll
    for (int s4 = 0; s4 < 4; ++s4) {
      const _Float16* vr = vb + (size_t)(s4 * 16 + lrow) * SS + jb + lkb;
      oacc[s4] = wmma32(pa, *(const v16h*)vr, oacc[s4]);
    }
  }

  int b = bh >> 3, h = bh & 7;
#pragma unroll
  for (int s4 = 0; s4 < 4; ++s4) {
#pragma unroll
    for (int r = 0; r < 8; ++r) {
      int srow = qt * 16 + r + 8 * hi;
      int d = s4 * 16 + lrow;
      o[((size_t)(b * SS + srow)) * EE + h * HDD + d] =
          (_Float16)(oacc[s4][r] / l[r]);
    }
  }
}

// ---------------------------------------------------------------------------
// Host orchestration
// ---------------------------------------------------------------------------
extern "C" void kernel_launch(void* const* d_in, const int* in_sizes, int n_in,
                              void* d_out, int out_size, void* d_ws,
                              size_t ws_size, hipStream_t stream) {
  (void)in_sizes; (void)n_in; (void)out_size; (void)ws_size;
  const float* x_in = (const float*)d_in[0];
  const float* ln1w = (const float*)d_in[1];
  const float* ln1b = (const float*)d_in[2];
  const float* ln2w = (const float*)d_in[3];
  const float* ln2b = (const float*)d_in[4];
  const float* qkvw = (const float*)d_in[5];
  const float* qkvb = (const float*)d_in[6];
  const float* outw = (const float*)d_in[7];
  const float* outb = (const float*)d_in[8];
  const float* fc1w = (const float*)d_in[9];
  const float* fc1b = (const float*)d_in[10];
  const float* fc2w = (const float*)d_in[11];
  const float* fc2b = (const float*)d_in[12];
  float* x = (float*)d_out;  // running residual stream

  size_t off = 0;
  auto alloc = [&](size_t bytes) -> char* {
    char* p = (char*)d_ws + off;
    off += (bytes + 255) & ~(size_t)255;
    return p;
  };
  _Float16* h16  = (_Float16*)alloc((size_t)MM * EE * 2);
  _Float16* q16  = (_Float16*)alloc((size_t)MM * EE * 2);
  _Float16* k16  = (_Float16*)alloc((size_t)MM * EE * 2);
  _Float16* v16t = (_Float16*)alloc((size_t)MM * EE * 2);
  _Float16* o16  = (_Float16*)alloc((size_t)MM * EE * 2);
  _Float16* ff16 = (_Float16*)alloc((size_t)MM * FFF * 2);
  _Float16* wq16 = (_Float16*)alloc((size_t)3 * EE * EE * 2);
  _Float16* wo16 = (_Float16*)alloc((size_t)EE * EE * 2);
  _Float16* wf1  = (_Float16*)alloc((size_t)FFF * EE * 2);
  _Float16* wf2  = (_Float16*)alloc((size_t)EE * FFF * 2);

  hipMemcpyAsync(x, x_in, (size_t)MM * EE * sizeof(float),
                 hipMemcpyDeviceToDevice, stream);

  for (int i = 0; i < 3; ++i) {
    // Convert this block's weights to f16
    cvt_f16_kernel<<<(3 * EE * EE) / 256, 256, 0, stream>>>(
        qkvw + (size_t)i * 3 * EE * EE, wq16, 3 * EE * EE);
    cvt_f16_kernel<<<(EE * EE) / 256, 256, 0, stream>>>(
        outw + (size_t)i * EE * EE, wo16, EE * EE);
    cvt_f16_kernel<<<(FFF * EE) / 256, 256, 0, stream>>>(
        fc1w + (size_t)i * FFF * EE, wf1, FFF * EE);
    cvt_f16_kernel<<<(EE * FFF) / 256, 256, 0, stream>>>(
        fc2w + (size_t)i * EE * FFF, wf2, EE * FFF);

    // h = LN1(x)
    ln_f16_kernel<<<MM, 256, 0, stream>>>(x, ln1w + i * EE, ln1b + i * EE, h16);

    // qkv = h @ qkv_w^T + b  -> q16/k16/v16t  (M/32)*(3E/64)=6144 waves
    gemm_wmma_kernel<EPI_QKV><<<768, 256, 0, stream>>>(
        h16, wq16, qkvb + (size_t)i * 3 * EE, nullptr, nullptr, nullptr,
        q16, k16, v16t, MM, 3 * EE, EE);

    // attention -> o16   (B*H*S/16 = 4096 waves)
    flash_attn_kernel<<<512, 256, 0, stream>>>(q16, k16, v16t, o16, i);

    // x = x + o @ out_w^T + b   (256*8=2048 waves)
    gemm_wmma_kernel<EPI_RESID><<<256, 256, 0, stream>>>(
        o16, wo16, outb + (size_t)i * EE, x, x, nullptr,
        nullptr, nullptr, nullptr, MM, EE, EE);

    // h = LN2(x)
    ln_f16_kernel<<<MM, 256, 0, stream>>>(x, ln2w + i * EE, ln2b + i * EE, h16);

    // ff = gelu(h @ fc1_w^T + b)   (256*32=8192 waves)
    gemm_wmma_kernel<EPI_GELU><<<1024, 256, 0, stream>>>(
        h16, wf1, fc1b + (size_t)i * FFF, nullptr, nullptr, ff16,
        nullptr, nullptr, nullptr, MM, FFF, EE);

    // x = x + ff @ fc2_w^T + b   (256*8=2048 waves)
    gemm_wmma_kernel<EPI_RESID><<<256, 256, 0, stream>>>(
        ff16, wf2, fc2b + (size_t)i * EE, x, x, nullptr,
        nullptr, nullptr, nullptr, MM, EE, FFF);
  }
}
